// MF_19164144074976
// MI455X (gfx1250) — compile-verified, matrix operand order checked by reading, not run on hardware
//
#include <hip/hip_runtime.h>
#include <hip/hip_bf16.h>

#define N_NODES 200000
#define N_EDGES 800000
#define N_GRAPHS 4096

typedef __attribute__((ext_vector_type(16))) __bf16 bf16x16;
typedef __attribute__((ext_vector_type(8)))  float  floatx8;

__device__ __forceinline__ unsigned short f2bf(float f) {
    union { float f; unsigned u; } c; c.f = f;
    unsigned u = c.u;
    u += 0x7FFFu + ((u >> 16) & 1u);   // round-to-nearest-even f32 -> bf16
    return (unsigned short)(u >> 16);
}

// ---------------------------------------------------------------- degree & bucketing
__global__ void k_deg(const int* __restrict__ dst, int* __restrict__ deg, int nE) {
    int i = blockIdx.x * blockDim.x + threadIdx.x;
    if (i < nE) atomicAdd(&deg[dst[i]], 1);
}

__global__ void k_clamp_count(int* __restrict__ deg, int* __restrict__ counts, int n) {
    int i = blockIdx.x * blockDim.x + threadIdx.x;
    if (i < n) {
        int d = deg[i]; d = d > 5 ? 5 : d; deg[i] = d;
        atomicAdd(&counts[d], 1);
    }
}

__global__ void k_prefix(const int* __restrict__ counts,
                         int* __restrict__ seg_base, int* __restrict__ tile_base) {
    if (blockIdx.x == 0 && threadIdx.x == 0) {
        int s = 0, t = 0;
        for (int d = 0; d < 6; ++d) {
            seg_base[d] = s; tile_base[d] = t;
            s += counts[d]; t += (counts[d] + 15) >> 4;
        }
        seg_base[6] = s; tile_base[6] = t;
    }
}

__global__ void k_fill(const int* __restrict__ deg, const int* __restrict__ seg_base,
                       int* __restrict__ cursor, int* __restrict__ idx, int n) {
    int i = blockIdx.x * blockDim.x + threadIdx.x;
    if (i < n) {
        int d = deg[i];
        int p = atomicAdd(&cursor[d], 1);
        idx[seg_base[d] + p] = i;
    }
}

// ---------------------------------------------------------------- weight pre-pack (f32 -> bf16, WMMA B layout)
// Packed layout: [d][kc][ntile][lane][16], lane<16 -> N=nt*16+lane, K runs {0..7,16..23};
// lane>=16 -> N=nt*16+lane-16, K runs {8..15,24..31}.
__global__ void k_pack_w(const float* __restrict__ W, unsigned short* __restrict__ Wp,
                         int K, int Nc) {
    int total = 6 * K * Nc;
    int p = blockIdx.x * blockDim.x + threadIdx.x;
    if (p >= total) return;
    int KC = K >> 5, NT = Nc >> 4;
    int j    = p & 15;
    int lane = (p >> 4) & 31;
    int rest = p >> 9;
    int nt = rest % NT; rest /= NT;
    int kc = rest % KC; int d = rest / KC;
    int half = lane >> 4;
    int n  = nt * 16 + (lane & 15);
    int kk = half ? ((j < 8) ? 8 + j : 16 + j)
                  : ((j < 8) ? j     : 8  + j);
    int k = kc * 32 + kk;
    Wp[p] = f2bf(W[((size_t)d * K + k) * Nc + n]);
}

// ---------------------------------------------------------------- edge scatter-add (msg[dst] += h[src])
__global__ void k_scatter(const float* __restrict__ h, const int* __restrict__ src,
                          const int* __restrict__ dst, float* __restrict__ msg,
                          int nE, int C) {
    int c4 = C >> 2;
    int i = blockIdx.x * blockDim.x + threadIdx.x;
    if (i >= nE * c4) return;
    int e = i / c4;
    int f = (i - e * c4) << 2;
    float4 v = *(const float4*)(h + (size_t)src[e] * C + f);
    float* m = msg + (size_t)dst[e] * C + f;
    atomicAdd(m + 0, v.x); atomicAdd(m + 1, v.y);
    atomicAdd(m + 2, v.z); atomicAdd(m + 3, v.w);
}

// ---------------------------------------------------------------- bucketed MFConv GEMM via bf16 WMMA
__global__ void __launch_bounds__(128) k_mfconv(
    const float* __restrict__ msg, const float* __restrict__ h,
    const unsigned short* __restrict__ Wlp, const unsigned short* __restrict__ Wrp,
    const float* __restrict__ bl,
    const int* __restrict__ idx, const int* __restrict__ seg_base,
    const int* __restrict__ tile_base,
    float* __restrict__ out, int C_in, int C_out)
{
    extern __shared__ char smem[];
    int* rows = (int*)smem;                                  // 16 ints
    unsigned short* As = (unsigned short*)(smem + 64);       // 16 x C_in bf16 (msg)
    unsigned short* Ah = As + 16 * C_in;                     // 16 x C_in bf16 (root)

    const int t = blockIdx.x;
    int tb[7];
#pragma unroll
    for (int i = 0; i < 7; ++i) tb[i] = tile_base[i];
    if (t >= tb[6]) return;                                  // block-uniform
    int d = 0;
#pragma unroll
    for (int i = 0; i < 5; ++i) if (t >= tb[i + 1]) d = i + 1;
    const int tile_in = t - tb[d];
    const int row0    = seg_base[d] + tile_in * 16;
    const int seg_end = seg_base[d + 1];

    const int tid = threadIdx.x;
    if (tid < 16) {
        int r = row0 + tid;
        rows[tid] = (r < seg_end) ? idx[r] : -1;
    }
    __syncthreads();

    // cooperative gather + f32->bf16 convert of 16-row A tiles
    const int c4 = C_in >> 2;
    for (int p = tid; p < 16 * c4; p += 128) {
        int row = p / c4;
        int col = (p - row * c4) << 2;
        int node = rows[row];
        float4 vm = make_float4(0.f, 0.f, 0.f, 0.f), vh = vm;
        if (node >= 0) {
            vm = *(const float4*)(msg + (size_t)node * C_in + col);
            vh = *(const float4*)(h   + (size_t)node * C_in + col);
        }
        unsigned short* am = As + row * C_in + col;
        unsigned short* ar = Ah + row * C_in + col;
        am[0] = f2bf(vm.x); am[1] = f2bf(vm.y); am[2] = f2bf(vm.z); am[3] = f2bf(vm.w);
        ar[0] = f2bf(vh.x); ar[1] = f2bf(vh.y); ar[2] = f2bf(vh.z); ar[3] = f2bf(vh.w);
    }
    __syncthreads();

    const int wave  = tid >> 5;
    const int lane  = tid & 31;
    const int half  = lane >> 4;
    const int nIdx  = lane & 15;
    const int ntile = blockIdx.y * 4 + wave;
    const int n0    = ntile * 16;

    const int KC = C_in >> 5;      // 32-deep K chunks
    const int NT = C_out >> 4;

    float bias = bl[d * C_out + n0 + nIdx];    // bias depends only on column
    floatx8 acc;
#pragma unroll
    for (int r = 0; r < 8; ++r) acc[r] = bias;

    const int arow = half ? (lane - 16) : lane;
    const int aoff = half ? 8 : 0;

    for (int kc = 0; kc < KC; ++kc) {
        union { bf16x16 v; uint4 q[2]; } am, ar, wl, wr;
        const unsigned short* asrc = As + arow * C_in + kc * 32 + aoff;
        const unsigned short* hsrc = Ah + arow * C_in + kc * 32 + aoff;
        am.q[0] = *(const uint4*)(asrc);
        am.q[1] = *(const uint4*)(asrc + 16);
        ar.q[0] = *(const uint4*)(hsrc);
        ar.q[1] = *(const uint4*)(hsrc + 16);
        size_t wb = ((((size_t)d * KC + kc) * NT + ntile) * 32 + lane) * 16;
        wl.q[0] = *(const uint4*)(Wlp + wb);
        wl.q[1] = *(const uint4*)(Wlp + wb + 8);
        wr.q[0] = *(const uint4*)(Wrp + wb);
        wr.q[1] = *(const uint4*)(Wrp + wb + 8);
        acc = __builtin_amdgcn_wmma_f32_16x16x32_bf16(false, am.v, false, wl.v,
                                                      (short)0, acc, false, false);
        acc = __builtin_amdgcn_wmma_f32_16x16x32_bf16(false, ar.v, false, wr.v,
                                                      (short)0, acc, false, false);
    }

#pragma unroll
    for (int r = 0; r < 8; ++r) {
        int m = r + (half ? 8 : 0);
        int node = rows[m];
        if (node >= 0) out[(size_t)node * C_out + n0 + nIdx] = acc[r];
    }
}

// ---------------------------------------------------------------- global add pool
__global__ void k_pool(const float* __restrict__ h, const int* __restrict__ batch,
                       float* __restrict__ pool, int n, int C) {
    int c4 = C >> 2;
    int i = blockIdx.x * blockDim.x + threadIdx.x;
    if (i >= n * c4) return;
    int node = i / c4;
    int f = (i - node * c4) << 2;
    float4 v = *(const float4*)(h + (size_t)node * C + f);
    float* p = pool + (size_t)batch[node] * C + f;
    atomicAdd(p + 0, v.x); atomicAdd(p + 1, v.y);
    atomicAdd(p + 2, v.z); atomicAdd(p + 3, v.w);
}

// ---------------------------------------------------------------- final linear (1024 -> 2) + softmax, wave per node
__global__ void __launch_bounds__(128) k_final(
    const float* __restrict__ h0, const float* __restrict__ h1,
    const float* __restrict__ h2, const float* __restrict__ h3,
    const float* __restrict__ pool, const int* __restrict__ batch,
    const float* __restrict__ Wf, const float* __restrict__ bf2,
    float* __restrict__ out, int n)
{
    int gt   = blockIdx.x * blockDim.x + threadIdx.x;
    int node = gt >> 5;
    int lane = gt & 31;
    if (node >= n) return;
    int g = batch[node];
    float a0 = 0.f, a1 = 0.f;
    for (int j = lane; j < 1024; j += 32) {     // segment boundaries are multiples of 32 -> uniform branch
        float v;
        if      (j < 128) v = h0[(size_t)node * 128 + j];
        else if (j < 256) v = h1[(size_t)node * 128 + (j - 128)];
        else if (j < 512) v = h2[(size_t)node * 256 + (j - 256)];
        else if (j < 768) v = h3[(size_t)node * 256 + (j - 512)];
        else              v = pool[(size_t)g   * 256 + (j - 768)];
        a0 += v * Wf[2 * j];
        a1 += v * Wf[2 * j + 1];
    }
#pragma unroll
    for (int s = 16; s; s >>= 1) { a0 += __shfl_xor(a0, s, 32); a1 += __shfl_xor(a1, s, 32); }
    if (lane == 0) {
        a0 += bf2[0]; a1 += bf2[1];
        float mx = fmaxf(a0, a1);
        float e0 = __expf(a0 - mx), e1 = __expf(a1 - mx);
        float inv = 1.f / (e0 + e1);
        out[(size_t)node * 2 + 0] = e0 * inv;
        out[(size_t)node * 2 + 1] = e1 * inv;
    }
}

// ---------------------------------------------------------------- host launcher
extern "C" void kernel_launch(void* const* d_in, const int* in_sizes, int n_in,
                              void* d_out, int out_size, void* d_ws, size_t ws_size,
                              hipStream_t stream)
{
    (void)in_sizes; (void)n_in; (void)out_size; (void)ws_size;
    const float* x     = (const float*)d_in[0];
    const int*   ei    = (const int*)d_in[1];
    const int*   src   = ei;
    const int*   dst   = ei + N_EDGES;
    const int*   batch = (const int*)d_in[2];
    const float* Wl[4]  = {(const float*)d_in[3], (const float*)d_in[6],
                           (const float*)d_in[9], (const float*)d_in[12]};
    const float* blv[4] = {(const float*)d_in[4], (const float*)d_in[7],
                           (const float*)d_in[10], (const float*)d_in[13]};
    const float* Wr[4]  = {(const float*)d_in[5], (const float*)d_in[8],
                           (const float*)d_in[11], (const float*)d_in[14]};
    const float* Wf  = (const float*)d_in[15];
    const float* bf2 = (const float*)d_in[16];

    char* p = (char*)d_ws;
    auto alloc = [&](size_t bytes) -> char* {
        char* r = p; p += (bytes + 255) & ~(size_t)255; return r;
    };
    int* deg       = (int*)alloc((size_t)N_NODES * 4);
    int* idx       = (int*)alloc((size_t)N_NODES * 4);
    int* counts    = (int*)alloc(64);          // counts[0..5], cursor at +8
    int* cursor    = counts + 8;
    int* seg_base  = (int*)alloc(64);          // seg_base[0..6], tile_base at +8
    int* tile_base = seg_base + 8;
    float* msg  = (float*)alloc((size_t)N_NODES * 256 * 4);
    float* h0   = (float*)alloc((size_t)N_NODES * 128 * 4);
    float* h1   = (float*)alloc((size_t)N_NODES * 128 * 4);
    float* h2   = (float*)alloc((size_t)N_NODES * 256 * 4);
    float* h3   = (float*)alloc((size_t)N_NODES * 256 * 4);
    float* pool = (float*)alloc((size_t)N_GRAPHS * 256 * 4);
    const int cins[4]  = {64, 128, 128, 256};
    const int couts[4] = {128, 128, 256, 256};
    unsigned short* Wlp[4]; unsigned short* Wrp[4];
    for (int l = 0; l < 4; ++l) {
        size_t b = (size_t)6 * cins[l] * couts[l] * 2;
        Wlp[l] = (unsigned short*)alloc(b);
        Wrp[l] = (unsigned short*)alloc(b);
    }

    hipMemsetAsync(deg, 0, (size_t)N_NODES * 4, stream);
    hipMemsetAsync(counts, 0, 64, stream);
    hipMemsetAsync(pool, 0, (size_t)N_GRAPHS * 256 * 4, stream);

    const int tpb = 256;
    k_deg<<<(N_EDGES + tpb - 1) / tpb, tpb, 0, stream>>>(dst, deg, N_EDGES);
    k_clamp_count<<<(N_NODES + tpb - 1) / tpb, tpb, 0, stream>>>(deg, counts, N_NODES);
    k_prefix<<<1, 32, 0, stream>>>(counts, seg_base, tile_base);
    k_fill<<<(N_NODES + tpb - 1) / tpb, tpb, 0, stream>>>(deg, seg_base, cursor, idx, N_NODES);

    for (int l = 0; l < 4; ++l) {
        int tot = 6 * cins[l] * couts[l];
        k_pack_w<<<(tot + tpb - 1) / tpb, tpb, 0, stream>>>(Wl[l], Wlp[l], cins[l], couts[l]);
        k_pack_w<<<(tot + tpb - 1) / tpb, tpb, 0, stream>>>(Wr[l], Wrp[l], cins[l], couts[l]);
    }

    const float* hin = x;
    float* houts[4] = {h0, h1, h2, h3};
    const int ntiles = (N_NODES + 15) / 16 + 6;   // upper bound on sum of per-bucket tiles
    for (int l = 0; l < 4; ++l) {
        int C_in = cins[l], C_out = couts[l];
        hipMemsetAsync(msg, 0, (size_t)N_NODES * C_in * 4, stream);
        int tot = N_EDGES * (C_in / 4);
        k_scatter<<<(tot + tpb - 1) / tpb, tpb, 0, stream>>>(hin, src, dst, msg, N_EDGES, C_in);
        dim3 grid(ntiles, C_out / 64);
        size_t lds = 64 + (size_t)2 * 16 * C_in * 2;
        k_mfconv<<<grid, 128, lds, stream>>>(msg, hin, Wlp[l], Wrp[l], blv[l],
                                             idx, seg_base, tile_base,
                                             houts[l], C_in, C_out);
        hin = houts[l];
    }

    int totp = N_NODES * (256 / 4);
    k_pool<<<(totp + tpb - 1) / tpb, tpb, 0, stream>>>(h3, batch, pool, N_NODES, 256);
    long tf = (long)N_NODES * 32;
    k_final<<<(int)((tf + 127) / 128), 128, 0, stream>>>(h0, h1, h2, h3, pool, batch,
                                                         Wf, bf2, (float*)d_out, N_NODES);
}
